// EstraNet_52398601011448
// MI455X (gfx1250) — matrix operand order repaired and checked by reading, not compiled, not run on hardware
//
#include <hip/hip_runtime.h>

// ---------------------------------------------------------------------------
// EstraNet forward for MI455X (gfx1250, wave32, WMMA).
// All GEMMs -> v_wmma_f32_16x16x32_bf16 with vectorized b128 fragment loads
// and software-pipelined (double-buffered, unroll-2) K loop:
//   A  [m][k] row-major, K contiguous         -> 2x 16B loads / fragment
//   BT [n][k] row-major (pre-transposed B)    -> 2x 16B loads / fragment
// NT template = N-tiles per wave sharing one A fragment (1/2/4 WMMAs per K).
// ---------------------------------------------------------------------------

typedef __attribute__((ext_vector_type(16))) __bf16 v16bf;
typedef __attribute__((ext_vector_type(8)))  __bf16 v8bf;
typedef __attribute__((ext_vector_type(8)))  float  v8f;

static __device__ inline __bf16 f2bf(float f) {
  unsigned u = __float_as_uint(f);
  u += 0x7FFFu + ((u >> 16) & 1u);          // round-to-nearest-even
  unsigned short h = (unsigned short)(u >> 16);
  __bf16 r;
  __builtin_memcpy(&r, &h, sizeof(r));
  return r;
}

static __device__ inline v16bf cat16(v8bf lo, v8bf hi) {
  return __builtin_shufflevector(lo, hi, 0, 1, 2, 3, 4, 5, 6, 7,
                                 8, 9, 10, 11, 12, 13, 14, 15);
}

// ---------------- conv stem: Conv1D(SAME) -> ReLU -> AvgPool(4) -------------
__global__ void conv_pool_kernel(const float* __restrict__ in,
                                 const float* __restrict__ w,
                                 const float* __restrict__ bias,
                                 float* __restrict__ out,
                                 int Bn, int Lin, int Cin, int Cout,
                                 int ksize, int pad) {
  int Lout = Lin >> 2;
  long idx = (long)blockIdx.x * blockDim.x + threadIdx.x;
  long total = (long)Bn * Lout * Cout;
  if (idx >= total) return;
  int co = (int)(idx % Cout);
  long t = idx / Cout;
  int lp = (int)(t % Lout);
  int b  = (int)(t / Lout);
  const float* xb = in + (long)b * Lin * Cin;
  float acc = 0.f;
  for (int p = 0; p < 4; ++p) {
    int l = lp * 4 + p;
    float s = bias[co];
    for (int tap = 0; tap < ksize; ++tap) {
      int li = l + tap - pad;
      if (li < 0 || li >= Lin) continue;
      const float* xr = xb + (long)li * Cin;
      const float* wr = w + (long)tap * Cin * Cout + co;
      for (int ci = 0; ci < Cin; ++ci)
        s += xr[ci] * wr[(long)ci * Cout];
    }
    acc += fmaxf(s, 0.f);
  }
  out[idx] = acc * 0.25f;
}

// ---------------- LayerCentering over D=256, emit bf16 ----------------------
__global__ __launch_bounds__(256)
void lc_bf16_kernel(const float* __restrict__ in, __bf16* __restrict__ out,
                    int rows) {
  int wave = threadIdx.x >> 5;
  int lane = threadIdx.x & 31;
  int row  = blockIdx.x * 8 + wave;
  if (row >= rows) return;
  const float* r = in + (long)row * 256;
  float vals[8];
  float s = 0.f;
#pragma unroll
  for (int i = 0; i < 8; ++i) { vals[i] = r[lane + i * 32]; s += vals[i]; }
#pragma unroll
  for (int off = 16; off; off >>= 1) s += __shfl_xor(s, off, 32);
  float mean = s * (1.f / 256.f);
  __bf16* orow = out + (long)row * 256;
#pragma unroll
  for (int i = 0; i < 8; ++i) orow[lane + i * 32] = f2bf(vals[i] - mean);
}

// ---------------- f32 -> bf16 transposed cast (weights) ---------------------
// in: [batch][K][N] f32 -> out: [batch][N][K] bf16
__global__ void cast_transpose_kernel(const float* __restrict__ in,
                                      __bf16* __restrict__ out,
                                      long batch, int K, int N) {
  long i = (long)blockIdx.x * blockDim.x + threadIdx.x;
  long total = batch * (long)K * N;
  if (i >= total) return;
  int n = (int)(i % N);
  long t = i / N;
  int k = (int)(t % K);
  long z = t / K;
  out[z * (long)K * N + (long)n * K + k] = f2bf(in[i]);
}

// ---------------- Fourier feature map + ALiBi-exp modulation ----------------
// proj_{q,k}: [BS, H=8, 64] f32
//   qt : [BS][H][128]          bf16 (row-major, K=m contiguous for o-GEMM)
//   ktT: [B][H][128][S]        bf16 (transposed, K=s contiguous for kv-GEMM)
__global__ void fourier_mod_kernel(const float* __restrict__ pq,
                                   const float* __restrict__ pk,
                                   const float* __restrict__ gamma,
                                   __bf16* __restrict__ qt,
                                   __bf16* __restrict__ ktT,
                                   long BS, int S) {
  long idx = (long)blockIdx.x * blockDim.x + threadIdx.x;
  long total = BS * 512;             // BS * H(8) * 64
  if (idx >= total) return;
  int j  = (int)(idx & 63);
  int hh = (int)((idx >> 6) & 7);
  long row = idx >> 9;
  long b = row / S;
  int  s = (int)(row % S);
  float g = gamma[hh];
  float slope = 2.0f - 0.03125f * (float)j;     // (D-4j)/D * 2
  float tinv = 1.0f / (float)(S - 1);
  float t1 = (float)s * tinv;                   // forward position
  float t2 = (float)(S - 1 - s) * tinv;         // reversed position
  float eq1 = __expf(g * t1 * slope);
  float eq2 = __expf(g * t2 * slope);
  float ek1 = __expf(-g * t1 * slope);
  float ek2 = __expf(-g * t2 * slope);
  const float inv_m = 0.08838834764831845f;     // 1/sqrt(128)
  float a = pq[idx];
  float c = pk[idx];
  long oq = row * 1024 + (long)hh * 128 + j;
  qt[oq]      = f2bf(__cosf(a) * inv_m * eq1);
  qt[oq + 64] = f2bf(__sinf(a) * inv_m * eq2);
  long ok = (((b * 8 + hh) * 128) + j) * (long)S + s;
  ktT[ok]                = f2bf(__cosf(c) * inv_m * ek1);
  ktT[ok + 64 * (long)S] = f2bf(__sinf(c) * inv_m * ek2);
}

// ---------------- batched WMMA GEMM (A row-major, B pre-transposed) ---------
// C[z][m,n] = sum_k A[z][m*sAr + k] * BT[z][n*sBr + k]
// batch offset = (z / bdiv)*s?b0 + (z % bdiv)*s?b1
// flags: 1=bias[n]  2=relu  4=residual(add existing Cf)  8=store bf16 into Cb
// K loop is double-buffered and unrolled 2x so the register allocator can
// ping-pong fragment buffers without cross-iteration copies.
template <int NT>
__global__ __launch_bounds__(256)
void wmma_gemm_kernel(const __bf16* __restrict__ A, long sAr, long sAb0,
                      long sAb1,
                      const __bf16* __restrict__ BT, long sBr, long sBb0,
                      long sBb1,
                      float* __restrict__ Cf, __bf16* __restrict__ Cb,
                      long sCr, long sCc, long sCb0, long sCb1,
                      const float* __restrict__ bias,
                      int M, int N, int K, int bdiv, int flags) {
  int z  = blockIdx.y;
  int mt = M >> 4, nt = N >> 4;
  int ng = nt / NT;                          // N-tile groups
  int wave = threadIdx.x >> 5;
  int g = blockIdx.x * 8 + wave;
  if (g >= mt * ng) return;                  // wave-uniform: EXEC all-ones
  int tm  = g / ng;
  int tnb = (g % ng) * NT;
  int lane = threadIdx.x & 31;
  int half = lane >> 4;                      // lane group 0-15 / 16-31
  int l    = lane & 15;

  long offA = (long)(z / bdiv) * sAb0 + (long)(z % bdiv) * sAb1;
  long offB = (long)(z / bdiv) * sBb0 + (long)(z % bdiv) * sBb1;
  long offC = (long)(z / bdiv) * sCb0 + (long)(z % bdiv) * sCb1;

  // A fragment row base: lanes 0-15 start K at +0, lanes 16-31 at +8.
  const __bf16* Ap = A + offA + (long)(tm * 16 + l) * sAr + half * 8;
  // B fragment column base: lanes 0-15 K at +0, lanes 16-31 at +8.
  const __bf16* Bp[NT];
#pragma unroll
  for (int t = 0; t < NT; ++t)
    Bp[t] = BT + offB + (long)((tnb + t) * 16 + l) * sBr + half * 8;

  v8f acc[NT];
#pragma unroll
  for (int t = 0; t < NT; ++t) acc[t] = (v8f){};

  // ---- prologue: load K-step 0 ----
  v8bf a0 = *(const v8bf*)(Ap);
  v8bf a1 = *(const v8bf*)(Ap + 16);
  v8bf b0[NT], b1[NT];
#pragma unroll
  for (int t = 0; t < NT; ++t) {
    b0[t] = *(const v8bf*)(Bp[t]);
    b1[t] = *(const v8bf*)(Bp[t] + 16);
  }

  // ---- pipelined main loop: prefetch next K-step before computing current.
#pragma unroll 2
  for (int k0 = 32; k0 < K; k0 += 32) {
    v16bf av = cat16(a0, a1);
    v16bf bvv[NT];
#pragma unroll
    for (int t = 0; t < NT; ++t) bvv[t] = cat16(b0[t], b1[t]);

    Ap += 32;
    a0 = *(const v8bf*)(Ap);
    a1 = *(const v8bf*)(Ap + 16);
#pragma unroll
    for (int t = 0; t < NT; ++t) {
      Bp[t] += 32;
      b0[t] = *(const v8bf*)(Bp[t]);
      b1[t] = *(const v8bf*)(Bp[t] + 16);
    }

#pragma unroll
    for (int t = 0; t < NT; ++t)
      acc[t] = __builtin_amdgcn_wmma_f32_16x16x32_bf16(
          false, av, false, bvv[t], (short)0, acc[t], false, false);
  }

  // ---- epilogue K-step ----
  {
    v16bf av = cat16(a0, a1);
#pragma unroll
    for (int t = 0; t < NT; ++t)
      acc[t] = __builtin_amdgcn_wmma_f32_16x16x32_bf16(
          false, av, false, cat16(b0[t], b1[t]), (short)0, acc[t], false,
          false);
  }

  // C/D layout: VGPR v -> row tm*16 + v + half*8 ; lane l -> col base + l
#pragma unroll
  for (int t = 0; t < NT; ++t) {
    int col = (tnb + t) * 16 + l;
    float bval = (flags & 1) ? bias[col] : 0.0f;
#pragma unroll
    for (int v = 0; v < 8; ++v) {
      int row = tm * 16 + v + half * 8;
      long idx = offC + (long)row * sCr + (long)col * sCc;
      float val = acc[t][v] + bval;
      if (flags & 2) val = fmaxf(val, 0.0f);
      if (flags & 8) {
        Cb[idx] = f2bf(val);
      } else {
        if (flags & 4) val += Cf[idx];
        Cf[idx] = val;
      }
    }
  }
}

// ---------------------------------------------------------------------------
extern "C" void kernel_launch(void* const* d_in, const int* in_sizes, int n_in,
                              void* d_out, int out_size, void* d_ws,
                              size_t ws_size, hipStream_t stream) {
  (void)in_sizes; (void)n_in; (void)out_size; (void)ws_size;

  const int B = 8, Lraw = 131072, S = 2048, D = 256, H = 8, NL = 2;
  const long BS = (long)B * S;          // 16384

  const float* x    = (const float*)d_in[0];
  const float* cw0  = (const float*)d_in[1];
  const float* cb0  = (const float*)d_in[2];
  const float* cw1  = (const float*)d_in[3];
  const float* cb1  = (const float*)d_in[4];
  const float* cw2  = (const float*)d_in[5];
  const float* cb2  = (const float*)d_in[6];
  const float* Wq   = (const float*)d_in[7];
  const float* bq   = (const float*)d_in[8];
  const float* Wk   = (const float*)d_in[9];
  const float* bk   = (const float*)d_in[10];
  const float* Wv   = (const float*)d_in[11];
  const float* bv   = (const float*)d_in[12];
  const float* Wo   = (const float*)d_in[13];
  const float* bo   = (const float*)d_in[14];
  const float* omega= (const float*)d_in[15];
  const float* gamma= (const float*)d_in[16];
  const float* W1   = (const float*)d_in[17];
  const float* b1   = (const float*)d_in[18];
  const float* W2   = (const float*)d_in[19];
  const float* b2   = (const float*)d_in[20];

  float* h = (float*)d_out;             // [B,S,D] f32 — residual stream

  // ---- workspace layout ----
  char* wp = (char*)d_ws;
  auto alloc = [&](size_t bytes) -> void* {
    void* r = (void*)wp;
    wp += (bytes + 255) & ~(size_t)255;
    return r;
  };
  float*  h0    = (float*) alloc((size_t)B * 32768 * 8 * 4);
  float*  h1    = (float*) alloc((size_t)B * 8192 * 16 * 4);
  __bf16* a_bf  = (__bf16*)alloc((size_t)BS * 256 * 2);
  __bf16* qb    = (__bf16*)alloc((size_t)BS * 256 * 2);
  __bf16* kb    = (__bf16*)alloc((size_t)BS * 256 * 2);
  __bf16* vT    = (__bf16*)alloc((size_t)B * 256 * S * 2);   // [b][h*32+e][s]
  float*  projq = (float*) alloc((size_t)BS * 512 * 4);
  float*  projk = (float*) alloc((size_t)BS * 512 * 4);
  __bf16* qt    = (__bf16*)alloc((size_t)BS * 1024 * 2);     // [b,s][h][128]
  __bf16* ktT   = (__bf16*)alloc((size_t)B * H * 128 * S * 2); // [b][h][m][s]
  __bf16* kvT   = (__bf16*)alloc((size_t)B * H * 32 * 128 * 2); // [b][h][e][m]
  __bf16* o_bf  = (__bf16*)alloc((size_t)BS * 256 * 2);
  __bf16* ff1   = (__bf16*)alloc((size_t)BS * 1024 * 2);
  __bf16* wqT   = (__bf16*)alloc((size_t)256 * 256 * 2);
  __bf16* wkT   = (__bf16*)alloc((size_t)256 * 256 * 2);
  __bf16* wvT   = (__bf16*)alloc((size_t)256 * 256 * 2);
  __bf16* woT   = (__bf16*)alloc((size_t)256 * 256 * 2);
  __bf16* w1T   = (__bf16*)alloc((size_t)1024 * 256 * 2);
  __bf16* w2T   = (__bf16*)alloc((size_t)256 * 1024 * 2);
  __bf16* omT   = (__bf16*)alloc((size_t)H * 64 * 32 * 2);   // [h][m][d]

  auto castT = [&](const float* src, __bf16* dst, long batch, int K, int N) {
    long n = batch * (long)K * N;
    cast_transpose_kernel<<<(int)((n + 255) / 256), 256, 0, stream>>>(
        src, dst, batch, K, N);
  };

  auto gemm = [&](const __bf16* A, long sAr, long sAb0, long sAb1,
                  const __bf16* BT, long sBr, long sBb0, long sBb1,
                  float* Cf, __bf16* Cb, long sCr, long sCc, long sCb0,
                  long sCb1, const float* bias, int M, int N, int K,
                  int batches, int bdiv, int flags) {
    int mt = M / 16, nt = N / 16;
    int NTsel = (nt % 4 == 0) ? 4 : (nt % 2 == 0) ? 2 : 1;
    int groups = mt * (nt / NTsel);
    dim3 grid((groups + 7) / 8, batches);
    if (NTsel == 4)
      wmma_gemm_kernel<4><<<grid, 256, 0, stream>>>(
          A, sAr, sAb0, sAb1, BT, sBr, sBb0, sBb1, Cf, Cb, sCr, sCc, sCb0,
          sCb1, bias, M, N, K, bdiv, flags);
    else if (NTsel == 2)
      wmma_gemm_kernel<2><<<grid, 256, 0, stream>>>(
          A, sAr, sAb0, sAb1, BT, sBr, sBb0, sBb1, Cf, Cb, sCr, sCc, sCb0,
          sCb1, bias, M, N, K, bdiv, flags);
    else
      wmma_gemm_kernel<1><<<grid, 256, 0, stream>>>(
          A, sAr, sAb0, sAb1, BT, sBr, sBb0, sBb1, Cf, Cb, sCr, sCc, sCb0,
          sCb1, bias, M, N, K, bdiv, flags);
  };

  // ---- conv stem ----
  {
    long t0 = (long)B * 32768 * 8;
    conv_pool_kernel<<<(int)((t0 + 255) / 256), 256, 0, stream>>>(
        x, cw0, cb0, h0, B, Lraw, 1, 8, 11, 5);
    long t1 = (long)B * 8192 * 16;
    conv_pool_kernel<<<(int)((t1 + 255) / 256), 256, 0, stream>>>(
        h0, cw1, cb1, h1, B, 32768, 8, 16, 3, 1);
    long t2 = (long)B * S * D;
    conv_pool_kernel<<<(int)((t2 + 255) / 256), 256, 0, stream>>>(
        h1, cw2, cb2, h, B, 8192, 16, D, 3, 1);
  }

  // ---- transformer layers ----
  for (int L = 0; L < NL; ++L) {
    long wOff  = (long)L * 256 * 256;
    long w1Off = (long)L * 256 * 1024;
    castT(Wq + wOff, wqT, 1, 256, 256);
    castT(Wk + wOff, wkT, 1, 256, 256);
    castT(Wv + wOff, wvT, 1, 256, 256);
    castT(Wo + wOff, woT, 1, 256, 256);
    castT(W1 + w1Off, w1T, 1, 256, 1024);   // -> [1024][256]
    castT(W2 + w1Off, w2T, 1, 1024, 256);   // -> [256][1024]
    castT(omega + (long)L * H * 32 * 64, omT, H, 32, 64);  // -> [h][64][32]

    // a_in = LC(h) -> bf16
    lc_bf16_kernel<<<(int)(BS / 8), 256, 0, stream>>>(h, a_bf, (int)BS);

    // q,k = a_in @ W + b   [BS,256] x [256,256] -> bf16 row-major
    gemm(a_bf, 256, 0, 0, wqT, 256, 0, 0,
         nullptr, qb, 256, 1, 0, 0, bq + (long)L * 256,
         (int)BS, 256, 256, 1, 1, 1 | 8);
    gemm(a_bf, 256, 0, 0, wkT, 256, 0, 0,
         nullptr, kb, 256, 1, 0, 0, bk + (long)L * 256,
         (int)BS, 256, 256, 1, 1, 1 | 8);
    // v: same GEMM but C stored transposed per batch b: vT[b][h*32+e][s]
    gemm(a_bf, 256, (long)S * 256, 0, wvT, 256, 0, 0,
         nullptr, vT, 1, S, (long)256 * S, 0, bv + (long)L * 256,
         S, 256, 256, B, 1, 1 | 8);

    // Fourier projection per head: [BS,32] x [32,64], batched over H=8
    gemm(qb, 256, 32, 0, omT, 32, (long)64 * 32, 0,
         projq, nullptr, 512, 1, 64, 0, nullptr,
         (int)BS, 64, 32, H, 1, 0);
    gemm(kb, 256, 32, 0, omT, 32, (long)64 * 32, 0,
         projk, nullptr, 512, 1, 64, 0, nullptr,
         (int)BS, 64, 32, H, 1, 0);

    // cos/sin + exp modulation -> qt [BS,H,128], ktT [b][h][128][S]
    {
      long tot = BS * 512;
      fourier_mod_kernel<<<(int)((tot + 255) / 256), 256, 0, stream>>>(
          projq, projk, gamma + (long)L * H, qt, ktT, BS, S);
    }

    // kv[b,h] = kt^T @ v : [128,S] x [S,32], batched z = b*H+h (64)
    //   A = ktT (K=s contiguous), BT = vT (rows e, K=s contiguous)
    //   C -> kvT[b][h][e][m] via transposed strides
    gemm(ktT, S, (long)128 * S, 0,
         vT, S, (long)32 * S, 0,
         nullptr, kvT, 1, 128, (long)32 * 128, 0,
         nullptr, 128, 32, S, B * H, 1, 8);

    // o[b,h] = qt @ kv : [S,128] x [128,32], batched z = b*H+h
    gemm(qt, 1024, (long)S * 1024, 128,
         kvT, 128, (long)H * 32 * 128, (long)32 * 128,
         nullptr, o_bf, 256, 1, (long)S * 256, 32,
         nullptr, S, 32, 128, B * H, H, 8);

    // h += o @ Wo + bo
    gemm(o_bf, 256, 0, 0, woT, 256, 0, 0,
         h, nullptr, 256, 1, 0, 0, bo + (long)L * 256,
         (int)BS, 256, 256, 1, 1, 1 | 4);

    // FF: f_in = LC(h); h += relu(f_in@W1+b1)@W2 + b2
    lc_bf16_kernel<<<(int)(BS / 8), 256, 0, stream>>>(h, a_bf, (int)BS);
    gemm(a_bf, 256, 0, 0, w1T, 256, 0, 0,
         nullptr, ff1, 1024, 1, 0, 0, b1 + (long)L * 1024,
         (int)BS, 1024, 256, 1, 1, 1 | 2 | 8);
    gemm(ff1, 1024, 0, 0, w2T, 1024, 0, 0,
         h, nullptr, 256, 1, 0, 0, b2 + (long)L * 256,
         (int)BS, 256, 1024, 1, 1, 1 | 4);
  }
}
